// LSTMFC_57174604644869
// MI455X (gfx1250) — compile-verified, matrix operand order checked by reading, not run on hardware
//
#include <hip/hip_runtime.h>
#include <hip/hip_bf16.h>

// ---------------- problem constants ----------------
#define BATCH 64
#define TLEN  512
#define INF   256
#define HID   1024
#define OUTF  256
#define G4    (4*HID)   // 4096
#define PFD   512       // prefetch distance in elements (1 KB)

// ---------------- types ----------------
typedef unsigned short u16;
typedef __attribute__((ext_vector_type(16))) __bf16 v16bf;
typedef __attribute__((ext_vector_type(8)))  float  v8f;
typedef __attribute__((ext_vector_type(8)))  u16    u16x8;

static __device__ __forceinline__ u16 f2bf(float f) {
    unsigned u = __float_as_uint(f);
    u += 0x7FFFu + ((u >> 16) & 1u);          // round-to-nearest-even
    return (u16)(u >> 16);
}

static __device__ __forceinline__ float sig(float x) {
    return 1.0f / (1.0f + __expf(-x));
}

// WGP-scope prefetch via inline CDNA5 asm. __builtin_prefetch can only reach
// SE/SYS scope (which only fill GL2, per ISA) — scope 0 (WGP,
// the default when no scope modifier is given) pulls the line into ALL cache
// levels, so the demand global_load_b128 hits WGP$ instead of L2.
static __device__ __forceinline__ void pf_wgp(const void* p) {
    asm volatile("global_prefetch_b8 %0, off" :: "v"(p));
}
#define PF(p) pf_wgp((const void*)(p))

// Load one 16x32 bf16 WMMA fragment for this lane.
// rowptr: start of this lane's row (A: row M = lane&15; B: row N = lane&15 of W,
// since B[k][n] = W[n][k]).  k0 must already include (lane>>4)*8.
static __device__ __forceinline__ v16bf load_frag16(const u16* __restrict__ rowptr, int k0) {
    union { u16x8 h[2]; v16bf v; } u;
    u.h[0] = *(const u16x8*)(rowptr + k0);        // elems 0..7  : K = k0..k0+7
    u.h[1] = *(const u16x8*)(rowptr + k0 + 16);   // elems 8..15 : K = k0+16..k0+23
    return u.v;
}

static __device__ __forceinline__ v8f splat8(float s) {
    v8f v = {s, s, s, s, s, s, s, s};
    return v;
}

#define WMMA_BF16(A, B, C) \
    __builtin_amdgcn_wmma_f32_16x16x32_bf16(false, (A), false, (B), (short)0, (C), false, false)

// Software-pipelined single-tile GEMM K-loop (K multiple of 64, K >= 128).
// Double-buffered fragments + WGP-scope prefetch at 1KB distance.
static __device__ __forceinline__ v8f gemm_k_pipelined(
    v8f acc, const u16* __restrict__ arow, const u16* __restrict__ brow,
    int kadj, int KTOT)
{
    v16bf a0 = load_frag16(arow, kadj);
    v16bf b0 = load_frag16(brow, kadj);
    #pragma unroll 4
    for (int kk = 0; kk < KTOT - 64; kk += 64) {
        PF(arow + kk + PFD);
        PF(brow + kk + PFD);
        v16bf a1 = load_frag16(arow, kk + 32 + kadj);
        v16bf b1 = load_frag16(brow, kk + 32 + kadj);
        acc = WMMA_BF16(a0, b0, acc);
        a0 = load_frag16(arow, kk + 64 + kadj);
        b0 = load_frag16(brow, kk + 64 + kadj);
        acc = WMMA_BF16(a1, b1, acc);
    }
    {   // peeled last 64
        v16bf a1 = load_frag16(arow, KTOT - 32 + kadj);
        v16bf b1 = load_frag16(brow, KTOT - 32 + kadj);
        acc = WMMA_BF16(a0, b0, acc);
        acc = WMMA_BF16(a1, b1, acc);
    }
    return acc;
}

// Pipelined quad-accumulator K-loop: one A stream shared by 4 B streams.
// [kbeg, kend), multiples of 64, width >= 128.
static __device__ __forceinline__ void quad_k(
    v8f& ai, v8f& af, v8f& ag, v8f& ao,
    const u16* __restrict__ ar,
    const u16* __restrict__ w0, const u16* __restrict__ w1,
    const u16* __restrict__ w2, const u16* __restrict__ w3,
    int kadj, int kbeg, int kend)
{
    v16bf a0 = load_frag16(ar, kbeg + kadj);
    v16bf p0 = load_frag16(w0, kbeg + kadj);
    v16bf p1 = load_frag16(w1, kbeg + kadj);
    v16bf p2 = load_frag16(w2, kbeg + kadj);
    v16bf p3 = load_frag16(w3, kbeg + kadj);
    #pragma unroll 2
    for (int kk = kbeg; kk < kend - 64; kk += 64) {
        PF(ar + kk + PFD);
        PF(w0 + kk + PFD);
        PF(w1 + kk + PFD);
        PF(w2 + kk + PFD);
        PF(w3 + kk + PFD);
        v16bf a1 = load_frag16(ar, kk + 32 + kadj);
        v16bf q0 = load_frag16(w0, kk + 32 + kadj);
        v16bf q1 = load_frag16(w1, kk + 32 + kadj);
        v16bf q2 = load_frag16(w2, kk + 32 + kadj);
        v16bf q3 = load_frag16(w3, kk + 32 + kadj);
        ai = WMMA_BF16(a0, p0, ai);
        af = WMMA_BF16(a0, p1, af);
        ag = WMMA_BF16(a0, p2, ag);
        ao = WMMA_BF16(a0, p3, ao);
        a0 = load_frag16(ar, kk + 64 + kadj);
        p0 = load_frag16(w0, kk + 64 + kadj);
        p1 = load_frag16(w1, kk + 64 + kadj);
        p2 = load_frag16(w2, kk + 64 + kadj);
        p3 = load_frag16(w3, kk + 64 + kadj);
        ai = WMMA_BF16(a1, q0, ai);
        af = WMMA_BF16(a1, q1, af);
        ag = WMMA_BF16(a1, q2, ag);
        ao = WMMA_BF16(a1, q3, ao);
    }
    v16bf a1 = load_frag16(ar, kend - 32 + kadj);
    v16bf q0 = load_frag16(w0, kend - 32 + kadj);
    v16bf q1 = load_frag16(w1, kend - 32 + kadj);
    v16bf q2 = load_frag16(w2, kend - 32 + kadj);
    v16bf q3 = load_frag16(w3, kend - 32 + kadj);
    ai = WMMA_BF16(a0, p0, ai);
    af = WMMA_BF16(a0, p1, af);
    ag = WMMA_BF16(a0, p2, ag);
    ao = WMMA_BF16(a0, p3, ao);
    ai = WMMA_BF16(a1, q0, ai);
    af = WMMA_BF16(a1, q1, af);
    ag = WMMA_BF16(a1, q2, ag);
    ao = WMMA_BF16(a1, q3, ao);
}

// ---------------- prep kernels ----------------
__global__ void k_cvt_bf16(const float* __restrict__ src, u16* __restrict__ dst, int n) {
    int i = blockIdx.x * blockDim.x + threadIdx.x;
    if (i < n) dst[i] = f2bf(src[i]);
}

__global__ void k_bias_comb(const float* __restrict__ b_ih, const float* __restrict__ b_hh,
                            float* __restrict__ bias, int n) {
    int i = blockIdx.x * blockDim.x + threadIdx.x;
    if (i < n) bias[i] = b_ih[i] + b_hh[i];
}

__global__ void k_zero_u32(unsigned* __restrict__ p, int n) {
    int i = blockIdx.x * blockDim.x + threadIdx.x;
    if (i < n) p[i] = 0u;
}

// One-time NT GEMM: C(M,N) = A(M,K) @ B(N,K)^T, bf16 in/out, f32 accumulate.
// Used to fold the silence matrix into the recurrent/output weights:
//   W_eff[n][k] = sum_j W[n][j] * S[k][j]  (A = W rows, B = S rows).
__global__ __launch_bounds__(256) void k_gemm_nt_bf16(
    const u16* __restrict__ A, const u16* __restrict__ B, u16* __restrict__ C,
    int N, int K)
{
    const int lane = threadIdx.x & 31;
    const int w    = (blockIdx.x * blockDim.x + threadIdx.x) >> 5;
    const int ntiles = N >> 4;
    const int nt = w % ntiles, mt = w / ntiles;
    const int m0 = mt * 16, n0 = nt * 16;
    const int nr   = lane & 15;
    const int kadj = (lane >> 4) * 8;

    v8f acc = gemm_k_pipelined(splat8(0.0f),
                               A + (size_t)(m0 + nr) * K,
                               B + (size_t)(n0 + nr) * K, kadj, K);
    #pragma unroll
    for (int v = 0; v < 8; ++v) {
        const int row = m0 + (lane >> 4) * 8 + v;
        C[(size_t)row * N + n0 + nr] = f2bf(acc[v]);
    }
}

// ---------------- recurrent step kernel: gates + cell update ----------------
// gates(64,4096) = x_t @ W_ih^T + h_raw @ W_hh_eff^T + bias, then cell math.
// Split-K=2: each 4-gate tile group owned by a wave PAIR (same block).
//   ksplit0: x-part (K=256) + h K in [0,384)   -> 20 WMMA groups + bias
//   ksplit1: h K in [384,1024)                 -> 20 WMMA groups
// Partials combined through LDS; even wave does activations + c/h stores.
__global__ __launch_bounds__(256) void k_lstm_gates(
    const u16* __restrict__ xbf, const u16* __restrict__ Wih,
    const u16* __restrict__ Whh, const float* __restrict__ bias,
    const u16* __restrict__ h_in, float* __restrict__ c,
    u16* __restrict__ h_out, int t)
{
    __shared__ float red[4 * 1024];    // 4 wave-pairs per block x 1024 floats

    const int lane   = threadIdx.x & 31;
    const int waveid = threadIdx.x >> 5;      // 0..7
    const int ksplit = waveid & 1;
    const int pair   = waveid >> 1;           // 0..3
    const int tg     = blockIdx.x * 4 + pair; // 0..255 tile groups
    const int mt = tg & 3, jt = tg >> 2;
    const int m0 = mt * 16, j0 = jt * 16;
    const int nr   = lane & 15;
    const int half = lane >> 4;
    const int kadj = half * 8;

    v8f acc_i, acc_f, acc_g, acc_o;

    const u16* hrow = h_in + (size_t)(m0 + nr) * HID;
    const u16* wh0  = Whh + (size_t)(0 * HID + j0 + nr) * HID;
    const u16* wh1  = Whh + (size_t)(1 * HID + j0 + nr) * HID;
    const u16* wh2  = Whh + (size_t)(2 * HID + j0 + nr) * HID;
    const u16* wh3  = Whh + (size_t)(3 * HID + j0 + nr) * HID;

    if (ksplit == 0) {
        acc_i = splat8(bias[0 * HID + j0 + nr]);
        acc_f = splat8(bias[1 * HID + j0 + nr]);
        acc_g = splat8(bias[2 * HID + j0 + nr]);
        acc_o = splat8(bias[3 * HID + j0 + nr]);
        // x_t @ W_ih^T (K=256)
        const u16* xr  = xbf + (size_t)(m0 + nr) * TLEN * INF + (size_t)t * INF;
        const u16* wi0 = Wih + (size_t)(0 * HID + j0 + nr) * INF;
        const u16* wi1 = Wih + (size_t)(1 * HID + j0 + nr) * INF;
        const u16* wi2 = Wih + (size_t)(2 * HID + j0 + nr) * INF;
        const u16* wi3 = Wih + (size_t)(3 * HID + j0 + nr) * INF;
        quad_k(acc_i, acc_f, acc_g, acc_o, xr, wi0, wi1, wi2, wi3, kadj, 0, INF);
        // h part, K in [0, 384)
        quad_k(acc_i, acc_f, acc_g, acc_o, hrow, wh0, wh1, wh2, wh3, kadj, 0, 384);
    } else {
        acc_i = splat8(0.0f); acc_f = splat8(0.0f);
        acc_g = splat8(0.0f); acc_o = splat8(0.0f);
        // h part, K in [384, 1024)
        quad_k(acc_i, acc_f, acc_g, acc_o, hrow, wh0, wh1, wh2, wh3, kadj, 384, HID);
        // publish partials
        float* base = red + pair * 1024 + lane * 8;
        #pragma unroll
        for (int v = 0; v < 8; ++v) {
            base[0 * 256 + v] = acc_i[v];
            base[1 * 256 + v] = acc_f[v];
            base[2 * 256 + v] = acc_g[v];
            base[3 * 256 + v] = acc_o[v];
        }
    }
    __syncthreads();
    if (ksplit == 0) {
        const float* base = red + pair * 1024 + lane * 8;
        #pragma unroll
        for (int v = 0; v < 8; ++v) {
            const int row = m0 + half * 8 + v;
            const int col = j0 + nr;
            const size_t idx = (size_t)row * HID + col;
            float iv = sig(acc_i[v] + base[0 * 256 + v]);
            float fv = sig(acc_f[v] + base[1 * 256 + v]);
            float gv = tanhf(acc_g[v] + base[2 * 256 + v]);
            float ov = sig(acc_o[v] + base[3 * 256 + v]);
            float cn = fv * c[idx] + iv * gv;
            c[idx] = cn;
            h_out[idx] = f2bf(ov * tanhf(cn));
        }
    }
}

// ---------------- final batched output projection ----------------
// out(B*T, OUT) = H_all(T*B, HID) @ W_out_eff^T + b_out — one big GEMM
// hoisted off the recurrent critical path (32768 waves).
__global__ __launch_bounds__(256) void k_outproj_all(
    const u16* __restrict__ h_all, const u16* __restrict__ Wout,
    const float* __restrict__ b_out, float* __restrict__ out)
{
    const int lane = threadIdx.x & 31;
    const int w    = (blockIdx.x * blockDim.x + threadIdx.x) >> 5;  // 0..32767
    const int nt = w & 15;            // OUT/16 = 16 column tiles
    const int mt = w >> 4;            // (T*B)/16 = 2048 row tiles
    const int r0 = mt * 16, n0 = nt * 16;
    const int nr   = lane & 15;
    const int kadj = (lane >> 4) * 8;

    v8f acc = gemm_k_pipelined(splat8(b_out[n0 + nr]),
                               h_all + (size_t)(r0 + nr) * HID,
                               Wout  + (size_t)(n0 + nr) * HID, kadj, HID);
    #pragma unroll
    for (int v = 0; v < 8; ++v) {
        const int r = r0 + (lane >> 4) * 8 + v;   // r = t*BATCH + b
        const int b = r & (BATCH - 1);
        const int t = r >> 6;                      // BATCH = 64
        out[(size_t)b * TLEN * OUTF + (size_t)t * OUTF + n0 + nr] = acc[v];
    }
}

// ---------------- host ----------------
extern "C" void kernel_launch(void* const* d_in, const int* in_sizes, int n_in,
                              void* d_out, int out_size, void* d_ws, size_t ws_size,
                              hipStream_t stream) {
    (void)in_sizes; (void)n_in; (void)out_size; (void)ws_size;

    const float* x      = (const float*)d_in[0];   // (B,T,IN)
    const float* W_ih   = (const float*)d_in[1];   // (4H,IN)
    const float* W_hh   = (const float*)d_in[2];   // (4H,HID)
    const float* b_ih   = (const float*)d_in[3];   // (4H)
    const float* b_hh   = (const float*)d_in[4];   // (4H)
    const float* W_out  = (const float*)d_in[5];   // (OUT,HID)
    const float* b_out  = (const float*)d_in[6];   // (OUT)
    const float* sil    = (const float*)d_in[7];   // (HID,HID)
    float*       out    = (float*)d_out;           // (B,T,OUT)

    char* ws = (char*)d_ws;
    size_t off = 0;
    u16*   x_bf    = (u16*)(ws + off); off += (size_t)BATCH * TLEN * INF * 2;  // 16 MB
    u16*   Wih_bf  = (u16*)(ws + off); off += (size_t)G4 * INF * 2;            //  2 MB
    u16*   Whh_bf  = (u16*)(ws + off); off += (size_t)G4 * HID * 2;            //  8 MB
    u16*   Wout_bf = (u16*)(ws + off); off += (size_t)OUTF * HID * 2;          // .5 MB
    u16*   S_bf    = (u16*)(ws + off); off += (size_t)HID * HID * 2;           //  2 MB
    u16*   WhhEff  = (u16*)(ws + off); off += (size_t)G4 * HID * 2;            //  8 MB
    u16*   WoutEff = (u16*)(ws + off); off += (size_t)OUTF * HID * 2;          // .5 MB
    float* bias    = (float*)(ws + off); off += (size_t)G4 * 4;
    float* c_st    = (float*)(ws + off); off += (size_t)BATCH * HID * 4;       // c state
    u16*   h_zero  = (u16*)(ws + off); off += (size_t)BATCH * HID * 2;         // zeroed h_{-1}
    u16*   h_all   = (u16*)(ws + off); off += (size_t)TLEN * BATCH * HID * 2;  // 64 MB history
    off += 4096;  // tail pad: keeps 1KB-ahead WGP-scope prefetches inside ws

    const int TPB = 256;
    // --- one-time prep: conversions ---
    { int n = BATCH * TLEN * INF; k_cvt_bf16<<<n / TPB, TPB, 0, stream>>>(x, x_bf, n); }
    { int n = G4 * INF;           k_cvt_bf16<<<n / TPB, TPB, 0, stream>>>(W_ih, Wih_bf, n); }
    { int n = G4 * HID;           k_cvt_bf16<<<n / TPB, TPB, 0, stream>>>(W_hh, Whh_bf, n); }
    { int n = OUTF * HID;         k_cvt_bf16<<<n / TPB, TPB, 0, stream>>>(W_out, Wout_bf, n); }
    { int n = HID * HID;          k_cvt_bf16<<<n / TPB, TPB, 0, stream>>>(sil, S_bf, n); }
    k_bias_comb<<<G4 / TPB, TPB, 0, stream>>>(b_ih, b_hh, bias, G4);
    { // zero c (f32) + h_zero (bf16): contiguous in workspace
      int nwords = (BATCH * HID * 4 + BATCH * HID * 2) / 4;
      k_zero_u32<<<(nwords + TPB - 1) / TPB, TPB, 0, stream>>>((unsigned*)c_st, nwords);
    }
    // --- one-time prep: fold silence into weights (WMMA GEMMs) ---
    // WhhEff(4096,1024) = W_hh @ S^T : (4096/16)*(1024/16) = 16384 waves
    k_gemm_nt_bf16<<<2048, TPB, 0, stream>>>(Whh_bf, S_bf, WhhEff, HID, HID);
    // WoutEff(256,1024) = W_out @ S^T : 1024 waves
    k_gemm_nt_bf16<<<128, TPB, 0, stream>>>(Wout_bf, S_bf, WoutEff, HID, HID);

    // --- recurrent critical path: ONE kernel per step (512 waves each) ---
    for (int t = 0; t < TLEN; ++t) {
        const u16* h_prev = (t == 0) ? h_zero : (h_all + (size_t)(t - 1) * BATCH * HID);
        k_lstm_gates<<<64, TPB, 0, stream>>>(x_bf, Wih_bf, WhhEff, bias,
                                             h_prev, c_st,
                                             h_all + (size_t)t * BATCH * HID, t);
    }

    // --- batched output projection (off the critical path) ---
    k_outproj_all<<<4096, TPB, 0, stream>>>(h_all, WoutEff, b_out, out);
}
